// GraphSAGE_49718541418975
// MI455X (gfx1250) — matrix operand order repaired and checked by reading, not compiled, not run on hardware
//
#include <hip/hip_runtime.h>
#include <cstdint>

#define N_NODES 50000
#define N_EDGES 1600000
#define FDIM 128
#define NUM_GRAPHS 64

typedef __attribute__((ext_vector_type(16))) __bf16 v16bf;
typedef __attribute__((ext_vector_type(8)))  float  v8f;

struct __attribute__((aligned(16))) Chunk { unsigned int x[4]; };
struct Chunk2 { Chunk lo, hi; };

static __device__ __forceinline__ v16bf make_frag(Chunk lo, Chunk hi) {
  Chunk2 t{lo, hi};
  return __builtin_bit_cast(v16bf, t);
}

// f32 -> bf16 round-to-nearest-even (bit trick)
static __device__ __forceinline__ unsigned short f2bf(float f) {
  unsigned int u = __float_as_uint(f);
  u += 0x7fffu + ((u >> 16) & 1u);
  return (unsigned short)(u >> 16);
}

// ---------------- degree ----------------
__global__ void deg_kernel(const int* __restrict__ dst, float* __restrict__ deg) {
  int e = blockIdx.x * blockDim.x + threadIdx.x;
  if (e < N_EDGES) atomicAdd(&deg[dst[e]], 1.0f);
}

// ---------------- weight convert + transpose: WT[n][k] = bf16(W[k][n]) ----------------
__global__ void wconv_kernel(const float* __restrict__ W, unsigned short* __restrict__ WT) {
  int k = threadIdx.x, n = blockIdx.x;
  WT[n * FDIM + k] = f2bf(W[k * FDIM + n]);
}

// ---------------- edge scatter: agg[dst] += h[src] (wave per edge, 4 floats/lane) ----------------
__global__ void scatter_kernel(const int* __restrict__ src, const int* __restrict__ dst,
                               const float* __restrict__ h, float* __restrict__ agg) {
  long long tid = (long long)blockIdx.x * blockDim.x + threadIdx.x;
  int e = (int)(tid >> 5);
  if (e >= N_EDGES) return;
  int c = ((int)tid & 31) * 4;
  const float4 v = *(const float4*)(h + (long long)src[e] * FDIM + c);
  float* p = agg + (long long)dst[e] * FDIM + c;
  atomicAdd(p + 0, v.x);
  atomicAdd(p + 1, v.y);
  atomicAdd(p + 2, v.z);
  atomicAdd(p + 3, v.w);
}

// ---------------- fused SAGE layer: out = relu((agg/deg)@Wl + h@Wr + b) ----------------
// 256 threads = 8 waves. Block covers 64 nodes x 128 cols. Wave w owns col tile w.
__global__ __launch_bounds__(256) void sage_gemm_kernel(
    const float* __restrict__ hin, const float* __restrict__ agg,
    const float* __restrict__ deg,
    const unsigned short* __restrict__ WlT, const unsigned short* __restrict__ WrT,
    const float* __restrict__ bias, float* __restrict__ out)
{
  __shared__ unsigned short sAgg[64 * FDIM];  // bf16, deg-scaled
  __shared__ unsigned short sH[64 * FDIM];    // bf16 root features
  const int tid  = threadIdx.x;
  const int wave = tid >> 5;
  const int lane = tid & 31;
  const int m0   = blockIdx.x * 64;

  // stage 64x128 activation tile into LDS, converting to bf16 (agg scaled by 1/deg)
  for (int i = tid; i < 64 * FDIM; i += 256) {
    int row = i >> 7, col = i & 127;
    int node = m0 + row;
    float a = 0.f, h = 0.f;
    if (node < N_NODES) {
      float inv = 1.0f / fmaxf(deg[node], 1.0f);
      a = agg[(long long)node * FDIM + col] * inv;
      h = hin[(long long)node * FDIM + col];
    }
    sAgg[i] = f2bf(a);
    sH[i]   = f2bf(h);
  }
  __syncthreads();

  const int nloc = lane & 15;
  const int hi16 = lane >> 4;
  const int ncol = wave * 16 + nloc;
  const int khA  = hi16 * 8;    // A frag K phase (ISA 16-bit A 16x32 layout)
  const int khB  = hi16 * 16;   // unused; B frags also index with khA (same K order as A)
  const float bn = bias[ncol];
  const unsigned short* wl = WlT + ncol * FDIM; // contiguous K per output col
  const unsigned short* wr = WrT + ncol * FDIM;

  for (int mt = 0; mt < 4; ++mt) {
    const unsigned short* ra = sAgg + (mt * 16 + nloc) * FDIM;
    const unsigned short* rh = sH   + (mt * 16 + nloc) * FDIM;
    v8f acc = {};
    #pragma unroll
    for (int ks = 0; ks < 4; ++ks) {           // K = 128 of agg path
      int kp = ks * 32;
      v16bf af = make_frag(*(const Chunk*)(ra + kp + khA),
                           *(const Chunk*)(ra + kp + 16 + khA));
      v16bf bf = make_frag(*(const Chunk*)(wl + kp + khA),
                           *(const Chunk*)(wl + kp + 16 + khA));
      acc = __builtin_amdgcn_wmma_f32_16x16x32_bf16(false, af, false, bf,
                                                    (short)0, acc, false, false);
    }
    #pragma unroll
    for (int ks = 0; ks < 4; ++ks) {           // K = 128 of root path
      int kp = ks * 32;
      v16bf af = make_frag(*(const Chunk*)(rh + kp + khA),
                           *(const Chunk*)(rh + kp + 16 + khA));
      v16bf bf = make_frag(*(const Chunk*)(wr + kp + khA),
                           *(const Chunk*)(wr + kp + 16 + khA));
      acc = __builtin_amdgcn_wmma_f32_16x16x32_bf16(false, af, false, bf,
                                                    (short)0, acc, false, false);
    }
    // C layout: element r -> row r + 8*(lane>=16), col = ncol
    #pragma unroll
    for (int r = 0; r < 8; ++r) {
      int node = m0 + mt * 16 + r + hi16 * 8;
      if (node < N_NODES)
        out[(long long)node * FDIM + ncol] = fmaxf(acc[r] + bn, 0.0f);
    }
  }
}

// ---------------- pooling: add / max (count separately); values >= 0 post-relu ----------------
__global__ void pool_kernel(const float* __restrict__ h, const int* __restrict__ batch,
                            float* __restrict__ addp, unsigned int* __restrict__ maxp,
                            float* __restrict__ cnt)
{
  long long tid = (long long)blockIdx.x * blockDim.x + threadIdx.x;
  int n = (int)(tid >> 5);
  if (n >= N_NODES) return;
  int c = ((int)tid & 31) * 4;
  int g = batch[n];
  const float4 v = *(const float4*)(h + (long long)n * FDIM + c);
  float* ap = addp + g * FDIM + c;
  atomicAdd(ap + 0, v.x);
  atomicAdd(ap + 1, v.y);
  atomicAdd(ap + 2, v.z);
  atomicAdd(ap + 3, v.w);
  unsigned int* mp = maxp + g * FDIM + c;
  atomicMax(mp + 0, __float_as_uint(v.x));
  atomicMax(mp + 1, __float_as_uint(v.y));
  atomicMax(mp + 2, __float_as_uint(v.z));
  atomicMax(mp + 3, __float_as_uint(v.w));
  if (c == 0) atomicAdd(&cnt[g], 1.0f);
}

// ---------------- head MLP + log_softmax (one block per graph) ----------------
__global__ __launch_bounds__(128) void mlp_kernel(
    const float* __restrict__ addp, const unsigned int* __restrict__ maxp,
    const float* __restrict__ cnt,
    const float* __restrict__ W1, const float* __restrict__ b1,
    const float* __restrict__ W2, const float* __restrict__ b2,
    float* __restrict__ out)
{
  __shared__ float gv[3 * FDIM];
  __shared__ float hv[FDIM];
  __shared__ float lg[2];
  int g = blockIdx.x, t = threadIdx.x;
  float c = fmaxf(cnt[g], 1.0f);
  for (int i = t; i < 3 * FDIM; i += 128) {
    float v;
    if (i < FDIM)            v = addp[g * FDIM + i];
    else if (i < 2 * FDIM)   v = addp[g * FDIM + (i - FDIM)] / c;
    else                     v = __uint_as_float(maxp[g * FDIM + (i - 2 * FDIM)]);
    gv[i] = v;
  }
  __syncthreads();
  float s = b1[t];
  for (int k = 0; k < 3 * FDIM; ++k) s += gv[k] * W1[k * FDIM + t];
  hv[t] = fmaxf(s, 0.0f);
  __syncthreads();
  if (t < 2) {
    float l = b2[t];
    for (int k = 0; k < FDIM; ++k) l += hv[k] * W2[k * 2 + t];
    lg[t] = l;
  }
  __syncthreads();
  if (t == 0) {
    float m = fmaxf(lg[0], lg[1]);
    float lse = m + logf(expf(lg[0] - m) + expf(lg[1] - m));
    out[g * 2 + 0] = lg[0] - lse;
    out[g * 2 + 1] = lg[1] - lse;
  }
}

extern "C" void kernel_launch(void* const* d_in, const int* in_sizes, int n_in,
                              void* d_out, int out_size, void* d_ws, size_t ws_size,
                              hipStream_t stream)
{
  (void)in_sizes; (void)n_in; (void)out_size; (void)ws_size;
  const float* x     = (const float*)d_in[0];
  const int*   ei    = (const int*)d_in[1];
  const int*   batch = (const int*)d_in[2];
  const float* Wl[3] = {(const float*)d_in[3], (const float*)d_in[6], (const float*)d_in[9]};
  const float* Wr[3] = {(const float*)d_in[4], (const float*)d_in[7], (const float*)d_in[10]};
  const float* bb[3] = {(const float*)d_in[5], (const float*)d_in[8], (const float*)d_in[11]};
  const float* W1 = (const float*)d_in[12];
  const float* b1 = (const float*)d_in[13];
  const float* W2 = (const float*)d_in[14];
  const float* b2 = (const float*)d_in[15];

  char* ws = (char*)d_ws;
  size_t off = 0;
  auto take = [&](size_t bytes) {
    char* p = ws + off;
    off = (off + bytes + 255) & ~(size_t)255;
    return p;
  };
  float*          agg  = (float*)take((size_t)N_NODES * FDIM * 4);
  float*          hA   = (float*)take((size_t)N_NODES * FDIM * 4);
  float*          hB   = (float*)take((size_t)N_NODES * FDIM * 4);
  float*          deg  = (float*)take((size_t)N_NODES * 4);
  unsigned short* WT[6];
  for (int i = 0; i < 6; ++i) WT[i] = (unsigned short*)take((size_t)FDIM * FDIM * 2);
  float*          addp = (float*)take((size_t)NUM_GRAPHS * FDIM * 4);
  unsigned int*   maxp = (unsigned int*)take((size_t)NUM_GRAPHS * FDIM * 4);
  float*          cnt  = (float*)take((size_t)NUM_GRAPHS * 4);

  const int* src = ei;
  const int* dst = ei + N_EDGES;

  hipMemsetAsync(deg,  0, (size_t)N_NODES * 4, stream);
  hipMemsetAsync(addp, 0, (size_t)NUM_GRAPHS * FDIM * 4, stream);
  hipMemsetAsync(maxp, 0, (size_t)NUM_GRAPHS * FDIM * 4, stream);
  hipMemsetAsync(cnt,  0, (size_t)NUM_GRAPHS * 4, stream);

  deg_kernel<<<(N_EDGES + 255) / 256, 256, 0, stream>>>(dst, deg);
  for (int l = 0; l < 3; ++l) {
    wconv_kernel<<<FDIM, FDIM, 0, stream>>>(Wl[l], WT[2 * l]);
    wconv_kernel<<<FDIM, FDIM, 0, stream>>>(Wr[l], WT[2 * l + 1]);
  }

  const float* hin = x;
  float* houts[3] = {hA, hB, hA};
  const long long sthreads = (long long)N_EDGES * 32;
  for (int l = 0; l < 3; ++l) {
    hipMemsetAsync(agg, 0, (size_t)N_NODES * FDIM * 4, stream);
    scatter_kernel<<<(unsigned)((sthreads + 255) / 256), 256, 0, stream>>>(src, dst, hin, agg);
    sage_gemm_kernel<<<(N_NODES + 63) / 64, 256, 0, stream>>>(
        hin, agg, deg, WT[2 * l], WT[2 * l + 1], bb[l], houts[l]);
    hin = houts[l];
  }

  const long long pthreads = (long long)N_NODES * 32;
  pool_kernel<<<(unsigned)((pthreads + 255) / 256), 256, 0, stream>>>(hin, batch, addp, maxp, cnt);
  mlp_kernel<<<NUM_GRAPHS, 128, 0, stream>>>(addp, maxp, cnt, W1, b1, W2, b2, (float*)d_out);
}
